// NonLocalHorizontalBlock_69750268887543
// MI455X (gfx1250) — matrix operand order repaired and checked by reading, not compiled, not run on hardware
//
#include <hip/hip_runtime.h>
#include <cstdint>

// ---------------------------------------------------------------------------
// Problem constants (b=4, c=256, nz=64, nx=256, groups=16)
// ---------------------------------------------------------------------------
#define BATCH 4
#define CH    256
#define NZ    64
#define NX    256
#define PLANE (NZ * NX)              // 16384
#define BSTRIDE ((size_t)CH * PLANE) // 4194304 elems per batch (NCHW)
#define ELEMS ((size_t)BATCH * CH * NZ * NX) // 16777216

typedef __attribute__((ext_vector_type(16))) __bf16 v16bf;
typedef __attribute__((ext_vector_type(8)))  float  v8f;

// ---------------------------------------------------------------------------
// Helpers
// ---------------------------------------------------------------------------
__device__ __forceinline__ uint16_t f2bf(float f) {
  uint32_t u = __float_as_uint(f);
  uint32_t r = u + 0x7FFFu + ((u >> 16) & 1u);   // round-to-nearest-even
  return (uint16_t)(r >> 16);
}

__device__ __forceinline__ v8f wmma_bf16(v16bf a, v16bf b, v8f c) {
  return __builtin_amdgcn_wmma_f32_16x16x32_bf16(
      /*neg_a=*/false, a, /*neg_b=*/false, b,
      /*c_mod=*/(short)0, c, /*reuse_a=*/false, /*reuse_b=*/false);
}

// A-matrix fragment (16 M x 32 K, bf16), element (m,k) at base[m*stride + k].
// ISA layout: lanes 0-15 -> M=lane, K in {kh..kh+7, 16+kh..16+kh+7}, kh=0;
// lanes 16-31 same M, kh=8.  Two contiguous 16B chunks per lane.
__device__ __forceinline__ v16bf load_frag_a(const uint16_t* base, int stride) {
  int lane = threadIdx.x & 31;
  const uint16_t* p = base + (lane & 15) * stride + ((lane & 16) ? 8 : 0);
  union { v16bf v; uint4 q[2]; } r;
  r.q[0] = *(const uint4*)(p);
  r.q[1] = *(const uint4*)(p + 16);
  return r.v;
}

// B-matrix fragment (32 K x 16 N, bf16), element (k,n) at base[n*stride + k].
// ISA layout: lanes 0-15 -> N=lane, K in {kh..kh+7, 16+kh..16+kh+7}, kh=0;
// lanes 16-31 same N, kh=8.  Two contiguous 16B chunks per lane.
__device__ __forceinline__ v16bf load_frag_b(const uint16_t* base, int stride) {
  int lane = threadIdx.x & 31;
  const uint16_t* p = base + (lane & 15) * stride + ((lane & 16) ? 8 : 0);
  union { v16bf v; uint4 q[2]; } r;
  r.q[0] = *(const uint4*)(p);
  r.q[1] = *(const uint4*)(p + 16);
  return r.v;
}

// ---------------------------------------------------------------------------
// 1) GroupNorm statistics: one block per (b, group), contiguous 262144 floats
// ---------------------------------------------------------------------------
__global__ __launch_bounds__(256) void gn_stats_kernel(
    const float* __restrict__ x, float* __restrict__ mean, float* __restrict__ rstd) {
  const int N = 16 * PLANE;  // 262144 elements per (b,g)
  const float* p = x + (size_t)blockIdx.x * N;
  float s = 0.f, ss = 0.f;
  for (int i = threadIdx.x; i < N; i += 256) {
    float v = p[i];
    s += v;
    ss += v * v;
  }
  #pragma unroll
  for (int off = 16; off > 0; off >>= 1) {
    s  += __shfl_xor(s,  off, 32);
    ss += __shfl_xor(ss, off, 32);
  }
  __shared__ float sh[16];
  int w = threadIdx.x >> 5, lane = threadIdx.x & 31;
  if (lane == 0) { sh[w] = s; sh[w + 8] = ss; }
  __syncthreads();
  if (threadIdx.x == 0) {
    float ts = 0.f, tss = 0.f;
    #pragma unroll
    for (int i = 0; i < 8; ++i) { ts += sh[i]; tss += sh[i + 8]; }
    float m = ts / (float)N;
    float var = tss / (float)N - m * m;
    mean[blockIdx.x] = m;
    rstd[blockIdx.x] = rsqrtf(var + 1e-6f);
  }
}

// ---------------------------------------------------------------------------
// 2) Pack conv weights to bf16: wq/wk/wv -> [dz][co][ci], pw -> [co][ci]
// ---------------------------------------------------------------------------
__global__ __launch_bounds__(256) void pack_w_kernel(
    const float* __restrict__ qw, const float* __restrict__ kw,
    const float* __restrict__ vw, const float* __restrict__ pw,
    uint16_t* __restrict__ wq, uint16_t* __restrict__ wk,
    uint16_t* __restrict__ wv, uint16_t* __restrict__ wp) {
  int gid = blockIdx.x * 256 + threadIdx.x;      // 0 .. 196607
  if (gid < CH * CH) wp[gid] = f2bf(pw[gid]);    // pw is (co,ci,1,1)
  if (gid < 3 * CH * CH) {
    int dz = gid >> 16;
    int rem = gid & 65535;
    int co = rem >> 8, ci = rem & 255;
    int src = (co * CH + ci) * 3 + dz;           // (co,ci,3,1)
    wq[gid] = f2bf(qw[src]);
    wk[gid] = f2bf(kw[src]);
    wv[gid] = f2bf(vw[src]);
  }
}

// ---------------------------------------------------------------------------
// 3) GroupNorm apply -> channels-last bf16 h_t[b][z][x][c]
// ---------------------------------------------------------------------------
__global__ __launch_bounds__(256) void gn_apply_kernel(
    const float* __restrict__ x, const float* __restrict__ gamma,
    const float* __restrict__ beta, const float* __restrict__ mean,
    const float* __restrict__ rstd, uint16_t* __restrict__ h_t) {
  size_t gid = (size_t)blockIdx.x * 256 + threadIdx.x;  // over ELEMS, c fastest
  int c = (int)(gid & 255);
  size_t sp = gid >> 8;                 // (b*NZ+z)*NX + x
  int xi = (int)(sp & 255);
  size_t zr = sp >> 8;                  // b*NZ + z
  int b = (int)(zr >> 6), z = (int)(zr & 63);
  float v = x[(size_t)b * BSTRIDE + (size_t)c * PLANE + (size_t)z * NX + xi];
  int bg = b * 16 + (c >> 4);
  float hv = (v - mean[bg]) * rstd[bg] * gamma[c] + beta[c];
  h_t[gid] = f2bf(hv);
}

// ---------------------------------------------------------------------------
// 4) 3x1 conv as WMMA GEMM.  out = W(dz) @ h(z+dz-1) summed over dz, + bias.
//    transposed=1 -> out[b][z][x][c] bf16 (q_t / k_t)
//    transposed=0 -> out[b][c][z][x] bf16 (v)
//    grid: (B*NZ, 4 co-tiles, 4 x-tiles), block 256 = 8 waves,
//    wave computes one 16(co) x 32(x) output piece.
// ---------------------------------------------------------------------------
__global__ __launch_bounds__(256) void conv3x1_kernel(
    const uint16_t* __restrict__ h_t, const uint16_t* __restrict__ wpk,
    const float* __restrict__ bias, uint16_t* __restrict__ out, int transposed) {
  int bz = blockIdx.x;
  int b = bz >> 6, z = bz & 63;
  int w = threadIdx.x >> 5;
  int co0 = blockIdx.y * 64 + (w & 3) * 16;
  int x0  = blockIdx.z * 64 + (w >> 2) * 32;

  v8f acc0 = {}, acc1 = {};
  for (int dz = 0; dz < 3; ++dz) {
    int zi = z + dz - 1;
    if (zi < 0 || zi >= NZ) continue;   // zero padding along z
    const uint16_t* wbase = wpk + dz * (CH * CH) + co0 * CH;
    const uint16_t* hbase = h_t + ((size_t)(b * NZ + zi) * NX) * CH;
    #pragma unroll
    for (int kb = 0; kb < 8; ++kb) {
      int k0 = kb * 32;
      v16bf af = load_frag_a(wbase + k0, CH);
      v16bf b0 = load_frag_b(hbase + (size_t)x0 * CH + k0, CH);
      v16bf b1 = load_frag_b(hbase + (size_t)(x0 + 16) * CH + k0, CH);
      acc0 = wmma_bf16(af, b0, acc0);
      acc1 = wmma_bf16(af, b1, acc1);
    }
  }

  int lane = threadIdx.x & 31;
  int n = lane & 15;
  int mh = (lane & 16) ? 8 : 0;
  if (transposed) {
    // out[((b*NZ+z)*NX + x)*C + co] : 8 contiguous bf16 per lane -> b128 store
    size_t rowb = (size_t)(b * NZ + z) * NX;
    #pragma unroll
    for (int half = 0; half < 2; ++half) {
      const v8f& acc = half ? acc1 : acc0;
      int xx = x0 + half * 16 + n;
      uint16_t* pp = out + (rowb + xx) * CH + co0 + mh;
      uint32_t pk[4];
      #pragma unroll
      for (int i = 0; i < 4; ++i) {
        uint16_t lo = f2bf(acc[2 * i]     + bias[co0 + mh + 2 * i]);
        uint16_t hi = f2bf(acc[2 * i + 1] + bias[co0 + mh + 2 * i + 1]);
        pk[i] = (uint32_t)lo | ((uint32_t)hi << 16);
      }
      uint4 u; u.x = pk[0]; u.y = pk[1]; u.z = pk[2]; u.w = pk[3];
      *(uint4*)pp = u;
    }
  } else {
    // out[b][co][z][x] : lanes 0-15 contiguous x
    size_t base = (size_t)b * BSTRIDE + (size_t)z * NX;
    #pragma unroll
    for (int half = 0; half < 2; ++half) {
      const v8f& acc = half ? acc1 : acc0;
      int xx = x0 + half * 16 + n;
      #pragma unroll
      for (int r = 0; r < 8; ++r) {
        int co = co0 + mh + r;
        out[base + (size_t)co * PLANE + xx] = f2bf(acc[r] + bias[co]);
      }
    }
  }
}

// f32 C/D fragment -> LDS row-major [m][n], scaled
__device__ __forceinline__ void store_frag_f32_lds(float* base, int stride,
                                                   v8f acc, float scale) {
  int lane = threadIdx.x & 31;
  int n = lane & 15;
  int mh = (lane & 16) ? 8 : 0;
  #pragma unroll
  for (int r = 0; r < 8; ++r) base[(mh + r) * stride + n] = acc[r] * scale;
}

// ---------------------------------------------------------------------------
// 5) Axial attention over width.  One block = 32 query columns of one (b,z).
//    S = (Q^T K)/16 via WMMA -> LDS f32 ; softmax ; P bf16 in LDS ;
//    A_t = P @ V^T via WMMA -> a_t[b][z][x][c] bf16.
//    grid: B*NZ*8 = 2048 blocks, 256 threads. LDS: 32KB + 16KB.
// ---------------------------------------------------------------------------
__global__ __launch_bounds__(256) void attn_kernel(
    const uint16_t* __restrict__ q_t, const uint16_t* __restrict__ k_t,
    const uint16_t* __restrict__ v_n, uint16_t* __restrict__ a_t) {
  __shared__ float    S[32 * 256];   // scores, f32
  __shared__ uint16_t P[32 * 256];   // softmax probs, bf16

  int blk = blockIdx.x;
  int tile = blk & 7;                // which 32-query strip
  int bz = blk >> 3;                 // b*NZ + z
  int b = bz >> 6, z = bz & 63;
  int qx0 = tile * 32;
  int w = threadIdx.x >> 5;
  int rf = w & 1;                    // query-row fragment 0/1
  int nbBase = (w >> 1) * 4;         // 4 column fragments per wave
  int lane = threadIdx.x & 31;

  // ---- GEMM1: S[qx, kx] = sum_c Q[c,qx] K[c,kx] * (1/16)
  const uint16_t* qbase = q_t + ((size_t)bz * NX + qx0 + rf * 16) * CH;
  const uint16_t* kbase = k_t + (size_t)bz * NX * CH;
  v16bf afrag[8];
  #pragma unroll
  for (int kb = 0; kb < 8; ++kb) afrag[kb] = load_frag_a(qbase + kb * 32, CH);
  #pragma unroll
  for (int j = 0; j < 4; ++j) {
    int nb = nbBase + j;
    v8f acc = {};
    #pragma unroll
    for (int kb = 0; kb < 8; ++kb) {
      v16bf bf = load_frag_b(kbase + (size_t)(nb * 16) * CH + kb * 32, CH);
      acc = wmma_bf16(afrag[kb], bf, acc);
    }
    store_frag_f32_lds(&S[(rf * 16) * 256 + nb * 16], 256, acc, 0.0625f);
  }
  __syncthreads();

  // ---- softmax over kx (256) : 4 rows per wave, 8 values per lane
  for (int rr = 0; rr < 4; ++rr) {
    int r = w * 4 + rr;
    float* row = &S[r * 256];
    float vals[8];
    float mx = -1e30f;
    #pragma unroll
    for (int i = 0; i < 8; ++i) { vals[i] = row[lane * 8 + i]; mx = fmaxf(mx, vals[i]); }
    #pragma unroll
    for (int off = 16; off > 0; off >>= 1) mx = fmaxf(mx, __shfl_xor(mx, off, 32));
    float sum = 0.f;
    #pragma unroll
    for (int i = 0; i < 8; ++i) { vals[i] = __expf(vals[i] - mx); sum += vals[i]; }
    #pragma unroll
    for (int off = 16; off > 0; off >>= 1) sum += __shfl_xor(sum, off, 32);
    float inv = 1.0f / sum;
    uint16_t* prow = &P[r * 256];
    #pragma unroll
    for (int i = 0; i < 8; ++i) prow[lane * 8 + i] = f2bf(vals[i] * inv);
  }
  __syncthreads();

  // ---- GEMM2: a[qx, c] = sum_kx P[qx,kx] V[c,kx]
  #pragma unroll
  for (int kb = 0; kb < 8; ++kb)
    afrag[kb] = load_frag_a(&P[(rf * 16) * 256] + kb * 32, 256);
  const uint16_t* vbase = v_n + (size_t)b * BSTRIDE + (size_t)z * NX;
  #pragma unroll
  for (int j = 0; j < 4; ++j) {
    int nb = nbBase + j;
    v8f acc = {};
    #pragma unroll
    for (int kb = 0; kb < 8; ++kb) {
      v16bf bf = load_frag_b(vbase + (size_t)(nb * 16) * PLANE + kb * 32, PLANE);
      acc = wmma_bf16(afrag[kb], bf, acc);
    }
    int n = lane & 15;
    int mh = (lane & 16) ? 8 : 0;
    int c = nb * 16 + n;
    #pragma unroll
    for (int r = 0; r < 8; ++r) {
      int qx = qx0 + rf * 16 + mh + r;
      a_t[((size_t)bz * NX + qx) * CH + c] = f2bf(acc[r]);
    }
  }
}

// ---------------------------------------------------------------------------
// 6) 1x1 projection + bias + residual, WMMA GEMM, f32 output (NCHW).
//    grid: (B*NZ, 4, 4), block 256.
// ---------------------------------------------------------------------------
__global__ __launch_bounds__(256) void proj_kernel(
    const float* __restrict__ xin, const uint16_t* __restrict__ a_t,
    const uint16_t* __restrict__ pwb, const float* __restrict__ pb,
    float* __restrict__ out) {
  int bz = blockIdx.x;
  int b = bz >> 6, z = bz & 63;
  int w = threadIdx.x >> 5;
  int co0 = blockIdx.y * 64 + (w & 3) * 16;
  int x0  = blockIdx.z * 64 + (w >> 2) * 32;
  const uint16_t* abase = a_t + (size_t)bz * NX * CH;

  v8f acc0 = {}, acc1 = {};
  #pragma unroll
  for (int kb = 0; kb < 8; ++kb) {
    int k0 = kb * 32;
    v16bf af = load_frag_a(pwb + co0 * CH + k0, CH);
    v16bf b0 = load_frag_b(abase + (size_t)x0 * CH + k0, CH);
    v16bf b1 = load_frag_b(abase + (size_t)(x0 + 16) * CH + k0, CH);
    acc0 = wmma_bf16(af, b0, acc0);
    acc1 = wmma_bf16(af, b1, acc1);
  }

  int lane = threadIdx.x & 31;
  int n = lane & 15;
  int mh = (lane & 16) ? 8 : 0;
  size_t base = (size_t)b * BSTRIDE + (size_t)z * NX;
  #pragma unroll
  for (int half = 0; half < 2; ++half) {
    const v8f& acc = half ? acc1 : acc0;
    int xx = x0 + half * 16 + n;
    #pragma unroll
    for (int r = 0; r < 8; ++r) {
      int co = co0 + mh + r;
      size_t idx = base + (size_t)co * PLANE + xx;
      out[idx] = xin[idx] + pb[co] + acc[r];
    }
  }
}

// ---------------------------------------------------------------------------
// Host: workspace layout (bytes); a_t aliases h_t (h dead after convs).
// Total scratch ~130 MB.
// ---------------------------------------------------------------------------
#define WS_HT   ((size_t)0)
#define WS_QT   ((size_t)33554432)
#define WS_KT   ((size_t)67108864)
#define WS_VN   ((size_t)100663296)
#define WS_AT   WS_HT
#define WS_WQ   ((size_t)134217728)
#define WS_WK   ((size_t)134610944)
#define WS_WV   ((size_t)135004160)
#define WS_WP   ((size_t)135397376)
#define WS_MEAN ((size_t)135528448)
#define WS_RSTD ((size_t)135528704)

extern "C" void kernel_launch(void* const* d_in, const int* in_sizes, int n_in,
                              void* d_out, int out_size, void* d_ws, size_t ws_size,
                              hipStream_t stream) {
  (void)in_sizes; (void)n_in; (void)out_size; (void)ws_size;
  const float* x    = (const float*)d_in[0];
  const float* gn_g = (const float*)d_in[1];
  const float* gn_b = (const float*)d_in[2];
  const float* qw   = (const float*)d_in[3];
  const float* qb   = (const float*)d_in[4];
  const float* kw   = (const float*)d_in[5];
  const float* kb   = (const float*)d_in[6];
  const float* vw   = (const float*)d_in[7];
  const float* vb   = (const float*)d_in[8];
  const float* pw   = (const float*)d_in[9];
  const float* pb   = (const float*)d_in[10];
  float* out = (float*)d_out;

  uint8_t* ws = (uint8_t*)d_ws;
  uint16_t* h_t  = (uint16_t*)(ws + WS_HT);
  uint16_t* q_t  = (uint16_t*)(ws + WS_QT);
  uint16_t* k_t  = (uint16_t*)(ws + WS_KT);
  uint16_t* v_n  = (uint16_t*)(ws + WS_VN);
  uint16_t* a_t  = (uint16_t*)(ws + WS_AT);
  uint16_t* wq   = (uint16_t*)(ws + WS_WQ);
  uint16_t* wk   = (uint16_t*)(ws + WS_WK);
  uint16_t* wv   = (uint16_t*)(ws + WS_WV);
  uint16_t* wp   = (uint16_t*)(ws + WS_WP);
  float* mean = (float*)(ws + WS_MEAN);
  float* rstd = (float*)(ws + WS_RSTD);

  gn_stats_kernel<<<BATCH * 16, 256, 0, stream>>>(x, mean, rstd);
  pack_w_kernel<<<768, 256, 0, stream>>>(qw, kw, vw, pw, wq, wk, wv, wp);
  gn_apply_kernel<<<(int)(ELEMS / 256), 256, 0, stream>>>(x, gn_g, gn_b, mean, rstd, h_t);

  dim3 cgrid(BATCH * NZ, 4, 4);
  conv3x1_kernel<<<cgrid, 256, 0, stream>>>(h_t, wq, qb, q_t, 1);
  conv3x1_kernel<<<cgrid, 256, 0, stream>>>(h_t, wk, kb, k_t, 1);
  conv3x1_kernel<<<cgrid, 256, 0, stream>>>(h_t, wv, vb, v_n, 0);

  attn_kernel<<<BATCH * NZ * 8, 256, 0, stream>>>(q_t, k_t, v_n, a_t);

  proj_kernel<<<cgrid, 256, 0, stream>>>(x, a_t, wp, pb, out);
}
